// Attention_3607772529228
// MI455X (gfx1250) — matrix operand order repaired and checked by reading, not compiled, not run on hardware
//
#include <hip/hip_runtime.h>
#include <hip/hip_bf16.h>
#include <math.h>

// Shapes from the reference
#define BB   64
#define SS   512
#define TT   32
#define H2   1024      // 2*hidden
#define ATT  512

// LDS staging geometry (bytes)
#define A_STAGE 9216            // 64 rows x 36 f32 (stride-padded)
#define B_STAGE 40960           // 512 cols x 40 bf16 (stride-padded)
#define BOFF    (2 * A_STAGE)   // B buffers start after the two A buffers
#define BETAOFF (2 * A_STAGE + 2 * B_STAGE)
#define LDS_TOTAL (BETAOFF + 256)   // + 64 beta floats = 100608 B

typedef __attribute__((ext_vector_type(16))) __bf16 v16bf;
typedef __attribute__((ext_vector_type(8)))  __bf16 v8bf;
typedef __attribute__((ext_vector_type(8)))  float  v8f;

#if __has_builtin(__builtin_amdgcn_tanhf)
#define FAST_TANH(x) __builtin_amdgcn_tanhf(x)
#else
#define FAST_TANH(x) tanhf(x)
#endif

// ---------------------------------------------------------------------------
// CDNA5 async global->LDS copy, 16B per lane (GVS addressing), ASYNCcnt.
// ---------------------------------------------------------------------------
__device__ __forceinline__ void async_b128(unsigned lds_addr, int goff,
                                           const void* sbase) {
    asm volatile("global_load_async_to_lds_b128 %0, %1, %2"
                 :: "v"(lds_addr), "v"(goff), "s"(sbase) : "memory");
}
__device__ __forceinline__ void wait_async0() {
    asm volatile("s_wait_asynccnt 0x0" ::: "memory");
}

// ---------------------------------------------------------------------------
// Kernel 1: ht_mean[b,d] = mean_t ht[b,t,d]
// ---------------------------------------------------------------------------
__global__ void ht_mean_kernel(const float* __restrict__ ht,
                               float* __restrict__ mean) {
    int idx = blockIdx.x * blockDim.x + threadIdx.x;   // over B*H2
    int b = idx >> 10;
    int d = idx & 1023;
    float s = 0.f;
#pragma unroll
    for (int t = 0; t < TT; ++t)
        s += ht[((size_t)(b * TT + t) << 10) + d];
    mean[idx] = s * (1.0f / TT);
}

// ---------------------------------------------------------------------------
// Kernel 2: W1a = w1_w[:, 0:1024] -> bf16, K-sliced layout [kk][col][32]
// (each K-step's 32KB slab contiguous for coalesced async staging)
// ---------------------------------------------------------------------------
__global__ void w1a_cvt_kernel(const float* __restrict__ w1,
                               __bf16* __restrict__ wb) {
    int idx = blockIdx.x * blockDim.x + threadIdx.x;   // dst-linear ATT*H2
    int kk  = idx >> 14;
    int col = (idx >> 5) & 511;
    int ki  = idx & 31;
    wb[idx] = (__bf16)w1[(size_t)col * 2048 + (kk << 5) + ki];
}

// ---------------------------------------------------------------------------
// Kernel 3: c[b,a] = w1_b[a] + W1b[a,:] . ht_mean[b,:]
// ---------------------------------------------------------------------------
__global__ void cbias_kernel(const float* __restrict__ w1,
                             const float* __restrict__ w1b,
                             const float* __restrict__ mean,
                             float* __restrict__ c) {
    int idx = blockIdx.x * blockDim.x + threadIdx.x;   // over B*ATT
    int b = idx >> 9;
    int a = idx & 511;
    const float4* wrow = (const float4*)(w1 + (size_t)a * 2048 + 1024);
    const float4* mrow = (const float4*)(mean + ((size_t)b << 10));
    float acc = 0.f;
    for (int k = 0; k < H2 / 4; ++k) {
        float4 w = wrow[k];
        float4 m = mrow[k];
        acc += w.x * m.x + w.y * m.y + w.z * m.z + w.w * m.w;
    }
    c[idx] = acc + w1b[a];
}

// ---------------------------------------------------------------------------
// Kernel 4 (main): beta[b,s] = sum_a u[a]*tanh( (W1a @ h_row)[a] + c[b,a] )
// 512 threads = 16 waves, wave grid 4 M-tiles x 4 N-groups.
// Double-buffered async LDS staging: one barrier per K-step; slice kk+1 is
// issued right after the barrier and flies while slice kk is consumed.
// ---------------------------------------------------------------------------
__global__ __launch_bounds__(512)
void beta_gemm_kernel(const float* __restrict__ h,
                      const __bf16* __restrict__ wb2,   // [32][512][32] bf16
                      const float* __restrict__ c,      // [B,ATT]
                      const float* __restrict__ u,      // [ATT]
                      float* __restrict__ beta) {       // [B,S]
    extern __shared__ char smem[];                      // LDS_TOTAL bytes
    float* lds_beta = (float*)(smem + BETAOFF);

    const int tid  = threadIdx.x;
    const int lane = tid & 31;
    const int wave = tid >> 5;
    const int wm   = wave >> 2;            // 0..3 : 16-row tile
    const int wn   = wave & 3;             // 0..3 : 128-col group
    const int b    = blockIdx.x >> 3;
    const int s0   = (blockIdx.x & 7) * 64;

    if (tid < 64) lds_beta[tid] = 0.f;

    const int hi  = (lane >= 16) ? 1 : 0;
    const int l15 = lane & 15;

    const unsigned smbase = (unsigned)(size_t)smem;    // LDS byte base

    // ---- staging maps (slice-invariant) ----
    const float* h_b = h + ((size_t)b << 19);          // b*S*H2
    const int arow = tid >> 3, aj = tid & 7;           // A: 1x16B chunk/thread
    const unsigned a_lds0 = smbase + arow * 144 + aj * 16;
    const int a_gbase = ((s0 + arow) << 12) + aj * 16; // + kk*128 bytes

    unsigned b_lds0[4];
    int      b_gbase[4];
#pragma unroll
    for (int j = 0; j < 4; ++j) {                      // B: 4x16B chunks/thread
        int cch = (j << 9) + tid;                      // chunk 0..2047
        int col = cch >> 2, q = cch & 3;
        b_lds0[j]  = smbase + BOFF + col * 80 + q * 16;
        b_gbase[j] = cch << 4;                         // + kk*32768 bytes
    }

    // ---- fragment read pointers (buffer 0) ----
    const char* ApB = smem + (wm * 16 + l15) * 144 + (hi ? 32 : 0);
    const char* BpB = smem + BOFF + (wn * 128 + l15) * 80 + (hi ? 16 : 0);

    v8f acc[8] = {};   // 8 N-tiles of 16x16 f32 accumulators

    // prologue: stage slice 0 into buffer 0
    async_b128(a_lds0, a_gbase, h_b);
#pragma unroll
    for (int j = 0; j < 4; ++j)
        async_b128(b_lds0[j], b_gbase[j], wb2);

    for (int kk = 0; kk < 32; ++kk) {
        const int sel  = kk & 1;
        wait_async0();          // own slice-kk writes done
        __syncthreads();        // all slice-kk data visible; prev reads done

        if (kk < 31) {          // stage slice kk+1 into the other buffer
            const int nsel = sel ^ 1;
            async_b128(a_lds0 + nsel * A_STAGE,
                       a_gbase + ((kk + 1) << 7), h_b);
#pragma unroll
            for (int j = 0; j < 4; ++j)
                async_b128(b_lds0[j] + nsel * B_STAGE,
                           b_gbase[j] + ((kk + 1) << 15), wb2);
        }

        // ---- consume slice kk ----
        const float*  Ap = (const float*)(ApB + sel * A_STAGE);
        const __bf16* Bp = (const __bf16*)(BpB + sel * B_STAGE);

        float4 f0 = *(const float4*)(Ap);
        float4 f1 = *(const float4*)(Ap + 4);
        float4 f2 = *(const float4*)(Ap + 16);
        float4 f3 = *(const float4*)(Ap + 20);
        v16bf afrag;
        afrag[0]  = (__bf16)f0.x; afrag[1]  = (__bf16)f0.y;
        afrag[2]  = (__bf16)f0.z; afrag[3]  = (__bf16)f0.w;
        afrag[4]  = (__bf16)f1.x; afrag[5]  = (__bf16)f1.y;
        afrag[6]  = (__bf16)f1.z; afrag[7]  = (__bf16)f1.w;
        afrag[8]  = (__bf16)f2.x; afrag[9]  = (__bf16)f2.y;
        afrag[10] = (__bf16)f2.z; afrag[11] = (__bf16)f2.w;
        afrag[12] = (__bf16)f3.x; afrag[13] = (__bf16)f3.y;
        afrag[14] = (__bf16)f3.z; afrag[15] = (__bf16)f3.w;

        // 8 N-tiles in 2 batches of 4: distinct regs -> pipelined ds loads
#pragma unroll
        for (int half = 0; half < 2; ++half) {
            v16bf bfr[4];
#pragma unroll
            for (int t = 0; t < 4; ++t) {
                const __bf16* bp = Bp + (half * 4 + t) * 640;   // 16 cols*40
                v8bf blo = *(const v8bf*)(bp);
                v8bf bhi = *(const v8bf*)(bp + 16);
                bfr[t] = __builtin_shufflevector(
                    blo, bhi, 0, 1, 2, 3, 4, 5, 6, 7,
                    8, 9, 10, 11, 12, 13, 14, 15);
            }
#pragma unroll
            for (int t = 0; t < 4; ++t)
                acc[half * 4 + t] = __builtin_amdgcn_wmma_f32_16x16x32_bf16(
                    false, afrag, false, bfr[t], (short)0,
                    acc[half * 4 + t], false, false);
        }
    }

    // Epilogue: tanh + dot with u, reduce over this wave's 128 columns.
    float rowsum[8];
#pragma unroll
    for (int v = 0; v < 8; ++v) rowsum[v] = 0.f;
#pragma unroll
    for (int nt = 0; nt < 8; ++nt) {
        int   ncol = wn * 128 + nt * 16 + l15;
        float cb   = c[(b << 9) + ncol];
        float uu   = u[ncol];
#pragma unroll
        for (int v = 0; v < 8; ++v)
            rowsum[v] += uu * FAST_TANH(acc[nt][v] + cb);
    }
#pragma unroll
    for (int off = 1; off < 16; off <<= 1) {   // xor<16 stays within lane half
#pragma unroll
        for (int v = 0; v < 8; ++v)
            rowsum[v] += __shfl_xor(rowsum[v], off, 32);
    }
    if (l15 == 0) {
        int rbase = wm * 16 + (hi ? 8 : 0);
#pragma unroll
        for (int v = 0; v < 8; ++v)
            atomicAdd(&lds_beta[rbase + v], rowsum[v]);  // ds_add_f32
    }
    __syncthreads();
    if (tid < 64)
        beta[(b << 9) + s0 + tid] = lds_beta[tid];
}

// ---------------------------------------------------------------------------
// Kernel 5: masked softmax over S, in-place on beta -> alpha
// ---------------------------------------------------------------------------
__global__ __launch_bounds__(512)
void softmax_kernel(const int* __restrict__ mask, float* __restrict__ beta) {
    __shared__ float red[512];
    int b = blockIdx.x;
    int s = threadIdx.x;
    float x = beta[(b << 9) + s];
    if (mask[(b << 9) + s] == 0) x = -1e20f;
    red[s] = x;
    __syncthreads();
    for (int off = 256; off > 0; off >>= 1) {
        if (s < off) red[s] = fmaxf(red[s], red[s + off]);
        __syncthreads();
    }
    float mx = red[0];
    __syncthreads();
    float e = expf(x - mx);
    red[s] = e;
    __syncthreads();
    for (int off = 256; off > 0; off >>= 1) {
        if (s < off) red[s] += red[s + off];
        __syncthreads();
    }
    beta[(b << 9) + s] = e / red[0];
}

// ---------------------------------------------------------------------------
// Kernel 6: s[b,d] = sum_s alpha[b,s] * h[b,s,d]   (coalesced over d)
// ---------------------------------------------------------------------------
__global__ void out_kernel(const float* __restrict__ h,
                           const float* __restrict__ alpha,
                           float* __restrict__ out) {
    int bb = blockIdx.x >> 2;
    int d  = ((blockIdx.x & 3) << 8) + threadIdx.x;
    const float* hp = h + ((size_t)bb << 19) + d;
    const float* ap = alpha + (bb << 9);
    float acc = 0.f;
    for (int s = 0; s < SS; ++s)
        acc += ap[s] * hp[(size_t)s << 10];
    out[(bb << 10) + d] = acc;
}

// ---------------------------------------------------------------------------
extern "C" void kernel_launch(void* const* d_in, const int* in_sizes, int n_in,
                              void* d_out, int out_size, void* d_ws, size_t ws_size,
                              hipStream_t stream) {
    (void)in_sizes; (void)n_in; (void)out_size; (void)ws_size;
    const float* h    = (const float*)d_in[0];   // [B,S,H2]
    const int*   mask = (const int*)  d_in[1];   // [B,S]
    const float* ht   = (const float*)d_in[2];   // [B,T,H2]
    const float* w1w  = (const float*)d_in[3];   // [ATT,2*H2]
    const float* w1b  = (const float*)d_in[4];   // [ATT]
    const float* uw   = (const float*)d_in[5];   // [1,ATT]
    float* out = (float*)d_out;                  // [B,H2]

    char* ws = (char*)d_ws;
    float*  wsMean = (float*) (ws);                 // 256 KB : B*H2 f32
    float*  wsC    = (float*) (ws + (256 << 10));   // 128 KB : B*ATT f32
    float*  wsBeta = (float*) (ws + (384 << 10));   // 128 KB : B*S  f32
    __bf16* wsB    = (__bf16*)(ws + (512 << 10));   // 1 MB   : [32][512][32] bf16

    ht_mean_kernel<<<(BB * H2) / 256, 256, 0, stream>>>(ht, wsMean);
    w1a_cvt_kernel<<<(ATT * H2) / 256, 256, 0, stream>>>(w1w, wsB);
    cbias_kernel  <<<(BB * ATT) / 256, 256, 0, stream>>>(w1w, w1b, wsMean, wsC);
    beta_gemm_kernel<<<BB * 8, 512, LDS_TOTAL, stream>>>(h, wsB, wsC, uw, wsBeta);
    softmax_kernel<<<BB, 512, 0, stream>>>(mask, wsBeta);
    out_kernel<<<BB * 4, 256, 0, stream>>>(h, wsBeta, out);
}
